// QLlamaMLP_39092792328541
// MI455X (gfx1250) — compile-verified, matrix operand order checked by reading, not run on hardware
//
#include <hip/hip_runtime.h>

// ---------------------------------------------------------------------------
// CDNA5 (gfx1250) SwiGLU MLP:  out = (silu(x @ Wg^T) * (x @ Wu^T)) @ Wd^T
// f32 -> bf16 convert, WMMA bf16 GEMMs (f32 accumulate), async global->LDS
// double-buffered staging (GLOBAL_LOAD_ASYNC_TO_LDS_B128, ASYNCcnt).
// ---------------------------------------------------------------------------

typedef __attribute__((ext_vector_type(16))) __bf16 v16bf;
typedef __attribute__((ext_vector_type(8)))  float  v8f;

#define MDIM 8192     // B*S tokens
#define HDIM 4096     // hidden
#define IDIM 11008    // intermediate

#define BM 128        // workgroup M tile
#define BN 64         // workgroup N tile
#define BK 64         // K step per LDS stage
#define AST 72        // LDS row stride (elems) for A tile (64 + 8 pad, 16B-aligned rows)
#define BST 72        // LDS row stride (elems) for B tiles

// ---- f32 -> bf16 round-to-nearest-even -------------------------------------
__device__ __forceinline__ unsigned short f32_to_bf16(float f) {
  unsigned int b = __float_as_uint(f);
  b += 0x7FFFu + ((b >> 16) & 1u);
  return (unsigned short)(b >> 16);
}

__global__ __launch_bounds__(256)
void cvt_f32_bf16(const float* __restrict__ src,
                  unsigned short* __restrict__ dst, long long n4) {
  long long i = (long long)blockIdx.x * blockDim.x + threadIdx.x;
  long long stride = (long long)gridDim.x * blockDim.x;
  const float4* s4 = (const float4*)src;
  for (; i < n4; i += stride) {
    float4 v = s4[i];
    unsigned int lo = (unsigned int)f32_to_bf16(v.x) | ((unsigned int)f32_to_bf16(v.y) << 16);
    unsigned int hi = (unsigned int)f32_to_bf16(v.z) | ((unsigned int)f32_to_bf16(v.w) << 16);
    ((uint2*)dst)[i] = make_uint2(lo, hi);
  }
}

// ---- async global->LDS 16B copy (per-lane), tracked by ASYNCcnt -------------
__device__ __forceinline__ void async_copy16(const unsigned short* smem_dst,
                                             const unsigned short* gsrc) {
  // low 32 bits of a flat LDS address are the LDS byte offset (ISA aperture map)
  unsigned lds_addr = (unsigned)(size_t)smem_dst;
  asm volatile("global_load_async_to_lds_b128 %0, %1, off"
               :
               : "v"(lds_addr), "v"(gsrc)
               : "memory");
}

__device__ __forceinline__ void wait_async() {
#if __has_builtin(__builtin_amdgcn_s_wait_asynccnt)
  __builtin_amdgcn_s_wait_asynccnt(0);
#else
  asm volatile("s_wait_asynccnt 0x0" ::: "memory");
#endif
}

// ---- LDS fragment loader per CDNA5 16-bit A/B WMMA layout -------------------
// lane L: row/col = L&15, k-base = (L>>4)*8.
// VGPR dwords 0..3 hold k = kb..kb+7, dwords 4..7 hold k = kb+16..kb+23.
__device__ __forceinline__ v16bf load_frag(const unsigned short* p) {
  union { uint4 u[2]; v16bf v; } f;
  const uint4* q = (const uint4*)p;      // 16B aligned by construction
  f.u[0] = q[0];
  f.u[1] = q[2];                          // +16 elements = +32 bytes
  return f.v;
}

__device__ __forceinline__ v8f wmma_bf16(v16bf a, v16bf b, v8f c) {
  return __builtin_amdgcn_wmma_f32_16x16x32_bf16(
      /*neg_a=*/false, a, /*neg_b=*/false, b,
      /*c_mod=*/(short)0, c, /*reuse_a=*/false, /*reuse_b=*/false);
}

// ---------------------------------------------------------------------------
// GEMM:  C[M,N] = A[M,K] * B[N,K]^T    (A, B row-major bf16, f32 accumulate)
// FUSED: two B operands (gate, up); epilogue silu(g)*u -> bf16 Cbf.
// else : single B; epilogue writes f32 Cf.
// 256 threads = 8 waves (wave32), tile BM x BN, waves in 4(M) x 2(N) grid,
// each wave owns 32x32 = 2x2 WMMA 16x16 tiles per output matrix.
// ---------------------------------------------------------------------------
template <bool FUSED>
__global__ __launch_bounds__(256)
void gemm_wmma(const unsigned short* __restrict__ A,
               const unsigned short* __restrict__ B0,
               const unsigned short* __restrict__ B1,
               unsigned short* __restrict__ Cbf,
               float* __restrict__ Cf,
               int K, int Nfull) {
  __shared__ unsigned short smA [2][BM * AST];
  __shared__ unsigned short smB0[2][BN * BST];
  __shared__ unsigned short smB1[2][BN * BST];

  const int tid   = threadIdx.x;
  const int lane  = tid & 31;
  const int wid   = tid >> 5;
  const int wm    = (wid >> 1) * 32;          // wave M offset: 0,32,64,96
  const int wn    = (wid & 1) * 32;           // wave N offset: 0,32
  const int lr    = lane & 15;                // row (A) / col (B) within 16-tile
  const int lkb   = (lane >> 4) * 8;          // per-lane k base

  const int mBase = blockIdx.y * BM;
  const int nBase = blockIdx.x * BN;
  const int nK    = K / BK;

  // Issue one stage of async global->LDS copies (no VGPR staging).
  auto issue_tile = [&](int kt, int buf) {
    const int k0 = kt * BK;
#pragma unroll
    for (int i = 0; i < 4; ++i) {             // A: 128 rows x 8 x 16B
      int idx = tid + i * 256;
      int row = idx >> 3, c = (idx & 7) * 8;
      async_copy16(&smA[buf][row * AST + c],
                   A + (size_t)(mBase + row) * K + k0 + c);
    }
#pragma unroll
    for (int i = 0; i < 2; ++i) {             // B: 64 rows x 8 x 16B (each)
      int idx = tid + i * 256;
      int row = idx >> 3, c = (idx & 7) * 8;
      async_copy16(&smB0[buf][row * BST + c],
                   B0 + (size_t)(nBase + row) * K + k0 + c);
      if (FUSED)
        async_copy16(&smB1[buf][row * BST + c],
                     B1 + (size_t)(nBase + row) * K + k0 + c);
    }
  };

  v8f acc0[2][2] = {};
  v8f acc1[2][2] = {};

  issue_tile(0, 0);
  wait_async();
  __syncthreads();

  for (int kt = 0; kt < nK; ++kt) {
    const int buf = kt & 1;
    if (kt + 1 < nK) issue_tile(kt + 1, buf ^ 1);   // async fill of next stage

    const unsigned short* a  = smA [buf];
    const unsigned short* b0 = smB0[buf];
    const unsigned short* b1 = smB1[buf];

#pragma unroll
    for (int ks = 0; ks < 2; ++ks) {          // two k-32 sub-steps per stage
      v16bf af[2], bg[2], bu[2];
#pragma unroll
      for (int i = 0; i < 2; ++i)
        af[i] = load_frag(a + (wm + i * 16 + lr) * AST + ks * 32 + lkb);
#pragma unroll
      for (int j = 0; j < 2; ++j) {
        bg[j] = load_frag(b0 + (wn + j * 16 + lr) * BST + ks * 32 + lkb);
        if (FUSED)
          bu[j] = load_frag(b1 + (wn + j * 16 + lr) * BST + ks * 32 + lkb);
      }
#pragma unroll
      for (int i = 0; i < 2; ++i)
#pragma unroll
        for (int j = 0; j < 2; ++j) {
          acc0[i][j] = wmma_bf16(af[i], bg[j], acc0[i][j]);
          if (FUSED) acc1[i][j] = wmma_bf16(af[i], bu[j], acc1[i][j]);
        }
    }

    if (kt + 1 < nK) {
      wait_async();                           // drain async fills of next stage
      __syncthreads();
    }
  }

  // Epilogue. C/D layout: lane: n = lane&15; dword e: m = e + 8*(lane>>4).
  const int cn = lane & 15;
  const int cm = 8 * (lane >> 4);
#pragma unroll
  for (int i = 0; i < 2; ++i)
#pragma unroll
    for (int j = 0; j < 2; ++j)
#pragma unroll
      for (int e = 0; e < 8; ++e) {
        const size_t m = (size_t)(mBase + wm + i * 16 + cm + e);
        const size_t n = (size_t)(nBase + wn + j * 16 + cn);
        if (FUSED) {
          float g = acc0[i][j][e];
          float u = acc1[i][j][e];
          float s = g / (1.0f + __expf(-g));  // silu
          Cbf[m * Nfull + n] = f32_to_bf16(s * u);
        } else {
          Cf[m * Nfull + n] = acc0[i][j][e];
        }
      }
}

// ---------------------------------------------------------------------------
extern "C" void kernel_launch(void* const* d_in, const int* in_sizes, int n_in,
                              void* d_out, int out_size, void* d_ws, size_t ws_size,
                              hipStream_t stream) {
  const float* x  = (const float*)d_in[0];   // [M, H]
  const float* wg = (const float*)d_in[1];   // [I, H]
  const float* wu = (const float*)d_in[2];   // [I, H]
  const float* wd = (const float*)d_in[3];   // [H, I]
  float* out = (float*)d_out;                // [M, H]

  const size_t nX = (size_t)MDIM * HDIM;     // 33.5M
  const size_t nW = (size_t)IDIM * HDIM;     // 45.1M

  unsigned short* xb    = (unsigned short*)d_ws;
  unsigned short* wgb   = xb  + nX;
  unsigned short* wub   = wgb + nW;
  unsigned short* wdb   = wub + nW;
  unsigned short* inter = wdb + nW;

  // f32 -> bf16 conversions (bandwidth-bound, ~60us total at 23.3 TB/s)
  cvt_f32_bf16<<<4096, 256, 0, stream>>>(x,  xb,  (long long)(nX / 4));
  cvt_f32_bf16<<<4096, 256, 0, stream>>>(wg, wgb, (long long)(nW / 4));
  cvt_f32_bf16<<<4096, 256, 0, stream>>>(wu, wub, (long long)(nW / 4));
  cvt_f32_bf16<<<4096, 256, 0, stream>>>(wd, wdb, (long long)(nW / 4));

  // Fused gate/up GEMM + SwiGLU -> inter (bf16): C[M,I]
  dim3 g1(IDIM / BN, MDIM / BM);             // (172, 64)
  gemm_wmma<true><<<g1, 256, 0, stream>>>(xb, wgb, wub, inter, nullptr,
                                          HDIM, IDIM);

  // Down projection -> out (f32): C[M,H]
  dim3 g2(HDIM / BN, MDIM / BM);             // (64, 64)
  gemm_wmma<false><<<g2, 256, 0, stream>>>(inter, wdb, nullptr, nullptr, out,
                                           IDIM, HDIM);
}
